// MONet_50156628082755
// MI455X (gfx1250) — matrix operand order repaired and checked by reading, not compiled, not run on hardware
//
#include <hip/hip_runtime.h>
#include <hip/hip_bf16.h>

// ---------------------------------------------------------------------------
// MONet GNN forward (GAT branch + GCN branch + MLP head) for gfx1250.
// Dense GEMMs run on bf16 WMMA (v_wmma_f32_16x16x32_bf16, fp32 accumulate);
// edge softmax / segment-sums are wave-per-(edge,head) atomic scatters.
// ---------------------------------------------------------------------------

#define N_NODES 20000
#define N_EDGES 320000

typedef __attribute__((ext_vector_type(16))) __bf16 v16bf;
typedef __attribute__((ext_vector_type(8)))  float  v8f;

__device__ __forceinline__ unsigned short f2bf(float x) {
    unsigned int u = __float_as_uint(x);
    unsigned int r = u + 0x7FFFu + ((u >> 16) & 1u);   // round-to-nearest-even
    return (unsigned short)(r >> 16);
}

// ---------------------------------------------------------------------------
// WMMA GEMM: C[M,Nc] = A[M,Kpad](bf16) * Bt[Npad,Kpad](bf16)^T  (+bias, act)
// One wave per 16x16 output tile. Kpad multiple of 32; zero-padded operands.
// A-lane layout (16x32 bf16): lane L holds row L%16, K in {b..b+7, b+16..b+23},
// b = 8*(L>=16).  B-lane layout (32x16): lane L holds col L%16, same K set as A.
// C layout: lane L, vgpr j -> row 8*(L>=16)+j, col L%16.
// ---------------------------------------------------------------------------
__global__ void gemm_wmma_bf16(const unsigned short* __restrict__ A,
                               const unsigned short* __restrict__ Bt,
                               float* __restrict__ C,
                               unsigned short* __restrict__ Cbf,
                               const float* __restrict__ bias,
                               int M, int Kpad, int Nc, int ldc, int ldbf,
                               int act, int ntiles) {
    int gid  = blockIdx.x * blockDim.x + threadIdx.x;
    int wid  = gid >> 5;
    int lane = threadIdx.x & 31;
    int mtiles = M >> 4;
    if (wid >= mtiles * ntiles) return;          // uniform per-wave exit
    int tm = wid / ntiles, tn = wid % ntiles;
    int half = lane >> 4;
    int l16  = lane & 15;

    const unsigned short* arow = A  + (size_t)(tm * 16 + l16) * Kpad;
    const unsigned short* brow = Bt + (size_t)(tn * 16 + l16) * Kpad;

    v8f acc = {};
    int ksteps = Kpad >> 5;
    for (int ks = 0; ks < ksteps; ++ks) {
        union { v16bf v; unsigned int u[8]; } a, b;
        const unsigned int* ap = (const unsigned int*)(arow + (ks * 32 + half * 8));
#pragma unroll
        for (int j = 0; j < 4; ++j) { a.u[j] = ap[j]; a.u[4 + j] = ap[8 + j]; }
        const unsigned int* bp = (const unsigned int*)(brow + (ks * 32 + half * 8));
#pragma unroll
        for (int j = 0; j < 4; ++j) { b.u[j] = bp[j]; b.u[4 + j] = bp[8 + j]; }
        acc = __builtin_amdgcn_wmma_f32_16x16x32_bf16(
            false, a.v, false, b.v, (short)0, acc, false, false);
    }

    int nc = tn * 16 + l16;
    bool ncok = nc < Nc;
    float bv = (bias != nullptr && ncok) ? bias[nc] : 0.0f;
#pragma unroll
    for (int j = 0; j < 8; ++j) {
        int mr = tm * 16 + half * 8 + j;
        if (mr < M && ncok) {
            float v = acc[j] + bv;
            if (act == 1) v = fmaxf(v, 0.0f);                 // relu
            else if (act == 2) v = v > 0.0f ? v : 0.25f * v;  // leaky (MLP)
            C[(size_t)mr * ldc + nc] = v;
            if (Cbf) Cbf[(size_t)mr * ldbf + nc] = f2bf(v);
        }
    }
}

// ---------------------------------------------------------------------------
// Elementwise / conversion / graph kernels
// ---------------------------------------------------------------------------
__global__ void k_cvt_bf16(const float* __restrict__ in, unsigned short* __restrict__ out,
                           int M, int K, int Kpad) {
    int i = blockIdx.x * blockDim.x + threadIdx.x;
    if (i >= M * Kpad) return;
    int m = i / Kpad, k = i - m * Kpad;
    out[i] = (k < K) ? f2bf(in[(size_t)m * K + k]) : (unsigned short)0;
}

__global__ void k_cvt_bf16_rowscale(const float* __restrict__ in, const float* __restrict__ s,
                                    unsigned short* __restrict__ out, int M, int K, int Kpad) {
    int i = blockIdx.x * blockDim.x + threadIdx.x;
    if (i >= M * Kpad) return;
    int m = i / Kpad, k = i - m * Kpad;
    out[i] = (k < K) ? f2bf(in[(size_t)m * K + k] * s[m]) : (unsigned short)0;
}

// weights [K,Ncols] f32 -> [Npad,Kpad] bf16 transposed, zero padded
__global__ void k_transpose_bf16(const float* __restrict__ in, unsigned short* __restrict__ out,
                                 int K, int Ncols, int Kpad, int Npad) {
    int i = blockIdx.x * blockDim.x + threadIdx.x;
    if (i >= Npad * Kpad) return;
    int n = i / Kpad, k = i - n * Kpad;
    out[i] = (n < Ncols && k < K) ? f2bf(in[(size_t)k * Ncols + n]) : (unsigned short)0;
}

__global__ void k_deg(const int* __restrict__ src, const int* __restrict__ dst,
                      float* dego, float* degi, int E) {
    int e = blockIdx.x * blockDim.x + threadIdx.x;
    if (e >= E) return;
    atomicAdd(&dego[src[e]], 1.0f);
    atomicAdd(&degi[dst[e]], 1.0f);
}

__global__ void k_rsqrt(const float* __restrict__ d, float* __restrict__ out, int n) {
    int i = blockIdx.x * blockDim.x + threadIdx.x;
    if (i < n) out[i] = rsqrtf(fmaxf(d[i], 1.0f));
}

// el[n,h] = sum_d h[n,h,d]*al[h,d]   er likewise
__global__ void k_att(const float* __restrict__ h, const float* __restrict__ al,
                      const float* __restrict__ ar, float* __restrict__ el,
                      float* __restrict__ er, int N, int H, int D) {
    int i = blockIdx.x * blockDim.x + threadIdx.x;
    if (i >= N * H) return;
    int n = i / H, hh = i - n * H;
    const float* hp  = h + (size_t)n * H * D + hh * D;
    const float* alp = al + hh * D;
    const float* arp = ar + hh * D;
    float sl = 0.0f, sr = 0.0f;
    for (int d = 0; d < D; ++d) { sl += hp[d] * alp[d]; sr += hp[d] * arp[d]; }
    el[i] = sl; er[i] = sr;
}

// ee = exp(leaky(el[src]+er[dst])); denom[dst] += ee   (max-shift dropped: softmax invariant)
__global__ void k_edge_att(const int* __restrict__ src, const int* __restrict__ dst,
                           const float* __restrict__ el, const float* __restrict__ er,
                           float* __restrict__ ee, float* __restrict__ denom,
                           int E, int H, float slope) {
    int i = blockIdx.x * blockDim.x + threadIdx.x;
    if (i >= E * H) return;
    int e = i / H, hh = i - e * H;
    float v = el[src[e] * H + hh] + er[dst[e] * H + hh];
    v = v > 0.0f ? v : slope * v;
    float x = __expf(v);
    ee[i] = x;
    atomicAdd(&denom[dst[e] * H + hh], x);
}

// rst[dst,h,:] += h1[src,h,:] * ee/denom[dst,h]  -- one wave per (edge,head)
__global__ void k_edge_agg_gat(const int* __restrict__ src, const int* __restrict__ dst,
                               const float* __restrict__ h, const float* __restrict__ ee,
                               const float* __restrict__ denom, float* __restrict__ rst,
                               int E, int H, int D) {
    int gid = blockIdx.x * blockDim.x + threadIdx.x;
    int wid = gid >> 5, lane = gid & 31;
    if (wid >= E * H) return;
    int e = wid / H, hh = wid - e * H;
    int s = src[e], d = dst[e];
    float coef = ee[wid] / fmaxf(denom[d * H + hh], 1e-9f);
    const float* hp = h + (size_t)s * H * D + hh * D;
    float* rp = rst + (size_t)d * H * D + hh * D;
    for (int j = lane; j < D; j += 32) atomicAdd(&rp[j], hp[j] * coef);
}

// agg[dst,:] += h[src,:]   -- one wave per edge
__global__ void k_edge_agg_gcn(const int* __restrict__ src, const int* __restrict__ dst,
                               const float* __restrict__ h, float* __restrict__ agg,
                               int E, int G) {
    int gid = blockIdx.x * blockDim.x + threadIdx.x;
    int wid = gid >> 5, lane = gid & 31;
    if (wid >= E) return;
    int s = src[wid], d = dst[wid];
    const float* hp = h + (size_t)s * G;
    float* rp = agg + (size_t)d * G;
    for (int j = lane; j < G; j += 32) atomicAdd(&rp[j], hp[j]);
}

// out = act(rst + xr + bias); optional f32 and bf16 (padded) outputs
__global__ void k_gat_fin(const float* __restrict__ rst, const float* __restrict__ xr,
                          const float* __restrict__ bias, float* __restrict__ outf,
                          unsigned short* __restrict__ outbf, int N, int HD, int HDpad,
                          int do_elu) {
    int i = blockIdx.x * blockDim.x + threadIdx.x;
    if (i >= N * HDpad) return;
    int n = i / HDpad, c = i - n * HDpad;
    if (c >= HD) { if (outbf) outbf[i] = 0; return; }
    float v = rst[(size_t)n * HD + c] + xr[(size_t)n * HD + c] + bias[c];
    if (do_elu) v = v > 0.0f ? v : (__expf(v) - 1.0f);
    if (outf)  outf[(size_t)n * HD + c] = v;
    if (outbf) outbf[i] = f2bf(v);
}

// y = act(agg*rsqrt(deg_i)+b); outbf = bf16(y*rsqrt(deg_o)) for next conv layer
__global__ void k_gcn_fin(const float* __restrict__ agg, const float* __restrict__ ri,
                          const float* __restrict__ ro, const float* __restrict__ bias,
                          float* __restrict__ outf, unsigned short* __restrict__ outbf,
                          int N, int G, int Gpad, int do_relu) {
    int i = blockIdx.x * blockDim.x + threadIdx.x;
    if (i >= N * Gpad) return;
    int n = i / Gpad, c = i - n * Gpad;
    if (c >= G) { if (outbf) outbf[i] = 0; return; }
    float v = agg[(size_t)n * G + c] * ri[n] + bias[c];
    if (do_relu) v = fmaxf(v, 0.0f);
    if (outf)  outf[(size_t)n * G + c] = v;
    if (outbf) outbf[i] = f2bf(v * ro[n]);
}

// z = concat(x_gat, x_gcn) -> bf16 [N,128]
__global__ void k_concat_bf(const float* __restrict__ a, const float* __restrict__ b,
                            unsigned short* __restrict__ out, int N) {
    int i = blockIdx.x * blockDim.x + threadIdx.x;
    if (i >= N * 128) return;
    int n = i >> 7, c = i & 127;
    float v = (c < 64) ? a[(size_t)n * 64 + c] : b[(size_t)n * 64 + (c - 64)];
    out[i] = f2bf(v);
}

// ---------------------------------------------------------------------------
// Host side
// ---------------------------------------------------------------------------
struct Bump {
    char* base; size_t off;
    void* take(size_t bytes) {
        size_t o = (off + 255) & ~(size_t)255;
        off = o + bytes;
        return (void*)(base + o);
    }
};

static inline dim3 g1(long n) { return dim3((unsigned)((n + 255) / 256)); }

static void run_gemm(const unsigned short* A, const unsigned short* Bt, float* C,
                     unsigned short* Cbf, const float* bias, int M, int Kpad, int Nc,
                     int ldc, int ldbf, int act, hipStream_t s) {
    int ntiles = (Nc + 15) / 16;
    long threads = (long)(M / 16) * ntiles * 32;
    gemm_wmma_bf16<<<g1(threads), 256, 0, s>>>(A, Bt, C, Cbf, bias,
                                               M, Kpad, Nc, ldc, ldbf, act, ntiles);
}

extern "C" void kernel_launch(void* const* d_in, const int* in_sizes, int n_in,
                              void* d_out, int out_size, void* d_ws, size_t ws_size,
                              hipStream_t stream) {
    const int N = N_NODES, E = N_EDGES;
    const int H1 = 5, D1 = 100, HD1 = 500, HD1p = 512;
    const int H2 = 1, D2 = 64;
    const int G1 = 300, G1p = 320, G2 = 100, G2p = 128, OUTD = 64;

    const float* x   = (const float*)d_in[0];
    const int*   src = (const int*)d_in[1];
    const int*   dst = (const int*)d_in[2];
    const float* w1  = (const float*)d_in[3];
    const float* al1 = (const float*)d_in[4];
    const float* ar1 = (const float*)d_in[5];
    const float* b1  = (const float*)d_in[6];
    const float* r1  = (const float*)d_in[7];
    const float* w2  = (const float*)d_in[8];
    const float* al2 = (const float*)d_in[9];
    const float* ar2 = (const float*)d_in[10];
    const float* b2  = (const float*)d_in[11];
    const float* r2  = (const float*)d_in[12];
    const float* gw1 = (const float*)d_in[13];
    const float* gb1 = (const float*)d_in[14];
    const float* gw2 = (const float*)d_in[15];
    const float* gb2 = (const float*)d_in[16];
    const float* gw3 = (const float*)d_in[17];
    const float* gb3 = (const float*)d_in[18];
    const float* f1w = (const float*)d_in[19];
    const float* f1b = (const float*)d_in[20];
    const float* f2w = (const float*)d_in[21];
    const float* f2b = (const float*)d_in[22];

    Bump ws{(char*)d_ws, 0};
    typedef unsigned short u16;
    // ---- weight staging (bf16, transposed, padded) ----
    u16* w1t  = (u16*)ws.take((size_t)512 * 128 * 2);
    u16* r1t  = (u16*)ws.take((size_t)512 * 128 * 2);
    u16* w2t  = (u16*)ws.take((size_t)64 * 512 * 2);
    u16* r2t  = (u16*)ws.take((size_t)64 * 512 * 2);
    u16* gw1t = (u16*)ws.take((size_t)304 * 128 * 2);
    u16* gw2t = (u16*)ws.take((size_t)112 * 320 * 2);
    u16* gw3t = (u16*)ws.take((size_t)64 * 128 * 2);
    u16* f1wt = (u16*)ws.take((size_t)64 * 128 * 2);
    u16* f2wt = (u16*)ws.take((size_t)16 * 64 * 2);
    // ---- persistent activations ----
    u16*   xbf   = (u16*)ws.take((size_t)N * 128 * 2);
    u16*   xnbf  = (u16*)ws.take((size_t)N * 128 * 2);
    u16*   h1bf  = (u16*)ws.take((size_t)N * HD1p * 2);
    float* el1   = (float*)ws.take((size_t)N * H1 * 4);
    float* er1   = (float*)ws.take((size_t)N * H1 * 4);
    float* den1  = (float*)ws.take((size_t)N * H1 * 4);
    float* ee1   = (float*)ws.take((size_t)E * H1 * 4);
    float* h2    = (float*)ws.take((size_t)N * 64 * 4);
    float* xr2   = (float*)ws.take((size_t)N * 64 * 4);
    float* rst2  = (float*)ws.take((size_t)N * 64 * 4);     // becomes x_gat
    float* el2   = (float*)ws.take((size_t)N * 4);
    float* er2   = (float*)ws.take((size_t)N * 4);
    float* den2  = (float*)ws.take((size_t)N * 4);
    float* ee2   = (float*)ws.take((size_t)E * 4);
    float* dego  = (float*)ws.take((size_t)N * 4);
    float* degi  = (float*)ws.take((size_t)N * 4);
    float* ro    = (float*)ws.take((size_t)N * 4);
    float* ri    = (float*)ws.take((size_t)N * 4);
    u16*   zbf   = (u16*)ws.take((size_t)N * 128 * 2);
    float* z1f   = (float*)ws.take((size_t)N * 64 * 4);
    u16*   z1bf  = (u16*)ws.take((size_t)N * 64 * 2);
    // ---- big region: GAT1 temps, then reused for GCN temps ----
    size_t mark = ws.off;
    float* h1   = (float*)ws.take((size_t)N * HD1 * 4);
    float* xr1  = (float*)ws.take((size_t)N * HD1 * 4);
    float* rst1 = (float*)ws.take((size_t)N * HD1 * 4);
    ws.off = mark;                                  // GCN reuses this region
    float* t1    = (float*)ws.take((size_t)N * G1 * 4);
    float* agg1  = (float*)ws.take((size_t)N * G1 * 4);
    u16*   g1bf  = (u16*)ws.take((size_t)N * G1p * 2);
    float* t2    = (float*)ws.take((size_t)N * G2 * 4);
    float* agg2  = (float*)ws.take((size_t)N * G2 * 4);
    u16*   g2bf  = (u16*)ws.take((size_t)N * G2p * 2);
    float* t3    = (float*)ws.take((size_t)N * OUTD * 4);
    float* xgcn  = (float*)ws.take((size_t)N * OUTD * 4);
    (void)ws_size; (void)n_in; (void)in_sizes; (void)out_size;

    // ---- 0. weight conversion ----
    k_transpose_bf16<<<g1(512 * 128), 256, 0, stream>>>(w1,  w1t, 128, 500, 128, 512);
    k_transpose_bf16<<<g1(512 * 128), 256, 0, stream>>>(r1,  r1t, 128, 500, 128, 512);
    k_transpose_bf16<<<g1(64 * 512),  256, 0, stream>>>(w2,  w2t, 500, 64, 512, 64);
    k_transpose_bf16<<<g1(64 * 512),  256, 0, stream>>>(r2,  r2t, 500, 64, 512, 64);
    k_transpose_bf16<<<g1(304 * 128), 256, 0, stream>>>(gw1, gw1t, 128, 300, 128, 304);
    k_transpose_bf16<<<g1(112 * 320), 256, 0, stream>>>(gw2, gw2t, 300, 100, 320, 112);
    k_transpose_bf16<<<g1(64 * 128),  256, 0, stream>>>(gw3, gw3t, 100, 64, 128, 64);
    k_transpose_bf16<<<g1(64 * 128),  256, 0, stream>>>(f1w, f1wt, 128, 64, 128, 64);
    k_transpose_bf16<<<g1(16 * 64),   256, 0, stream>>>(f2w, f2wt, 64, 2, 64, 16);
    k_cvt_bf16<<<g1((long)N * 128), 256, 0, stream>>>(x, xbf, N, 128, 128);

    // ---- degrees ----
    hipMemsetAsync(dego, 0, (size_t)N * 4, stream);
    hipMemsetAsync(degi, 0, (size_t)N * 4, stream);
    k_deg<<<g1(E), 256, 0, stream>>>(src, dst, dego, degi, E);
    k_rsqrt<<<g1(N), 256, 0, stream>>>(dego, ro, N);
    k_rsqrt<<<g1(N), 256, 0, stream>>>(degi, ri, N);
    k_cvt_bf16_rowscale<<<g1((long)N * 128), 256, 0, stream>>>(x, ro, xnbf, N, 128, 128);

    // ---- GAT layer 1 ----
    run_gemm(xbf, w1t, h1,  nullptr, nullptr, N, 128, HD1, HD1, 0, 0, stream);
    run_gemm(xbf, r1t, xr1, nullptr, nullptr, N, 128, HD1, HD1, 0, 0, stream);
    k_att<<<g1((long)N * H1), 256, 0, stream>>>(h1, al1, ar1, el1, er1, N, H1, D1);
    hipMemsetAsync(den1, 0, (size_t)N * H1 * 4, stream);
    hipMemsetAsync(rst1, 0, (size_t)N * HD1 * 4, stream);
    k_edge_att<<<g1((long)E * H1), 256, 0, stream>>>(src, dst, el1, er1, ee1, den1, E, H1, 0.2f);
    k_edge_agg_gat<<<g1((long)E * H1 * 32), 256, 0, stream>>>(src, dst, h1, ee1, den1, rst1, E, H1, D1);
    k_gat_fin<<<g1((long)N * HD1p), 256, 0, stream>>>(rst1, xr1, b1, nullptr, h1bf, N, HD1, HD1p, 1);

    // ---- GAT layer 2 (H=1, D=64) ----
    run_gemm(h1bf, w2t, h2,  nullptr, nullptr, N, 512, 64, 64, 0, 0, stream);
    run_gemm(h1bf, r2t, xr2, nullptr, nullptr, N, 512, 64, 64, 0, 0, stream);
    k_att<<<g1((long)N * H2), 256, 0, stream>>>(h2, al2, ar2, el2, er2, N, H2, D2);
    hipMemsetAsync(den2, 0, (size_t)N * 4, stream);
    hipMemsetAsync(rst2, 0, (size_t)N * 64 * 4, stream);
    k_edge_att<<<g1((long)E * H2), 256, 0, stream>>>(src, dst, el2, er2, ee2, den2, E, H2, 0.2f);
    k_edge_agg_gat<<<g1((long)E * H2 * 32), 256, 0, stream>>>(src, dst, h2, ee2, den2, rst2, E, H2, D2);
    k_gat_fin<<<g1((long)N * 64), 256, 0, stream>>>(rst2, xr2, b2, rst2, nullptr, N, 64, 64, 0);
    // rst2 == x_gat (mean over 1 head, no activation)

    // ---- GCN branch ----
    run_gemm(xnbf, gw1t, t1, nullptr, nullptr, N, 128, G1, G1, 0, 0, stream);
    hipMemsetAsync(agg1, 0, (size_t)N * G1 * 4, stream);
    k_edge_agg_gcn<<<g1((long)E * 32), 256, 0, stream>>>(src, dst, t1, agg1, E, G1);
    k_gcn_fin<<<g1((long)N * G1p), 256, 0, stream>>>(agg1, ri, ro, gb1, nullptr, g1bf, N, G1, G1p, 1);

    run_gemm(g1bf, gw2t, t2, nullptr, nullptr, N, G1p, G2, G2, 0, 0, stream);
    hipMemsetAsync(agg2, 0, (size_t)N * G2 * 4, stream);
    k_edge_agg_gcn<<<g1((long)E * 32), 256, 0, stream>>>(src, dst, t2, agg2, E, G2);
    k_gcn_fin<<<g1((long)N * G2p), 256, 0, stream>>>(agg2, ri, ro, gb2, nullptr, g2bf, N, G2, G2p, 1);

    run_gemm(g2bf, gw3t, t3, nullptr, nullptr, N, G2p, OUTD, OUTD, 0, 0, stream);
    hipMemsetAsync(xgcn, 0, (size_t)N * OUTD * 4, stream);
    k_edge_agg_gcn<<<g1((long)E * 32), 256, 0, stream>>>(src, dst, t3, xgcn, E, OUTD);
    k_gcn_fin<<<g1((long)N * OUTD), 256, 0, stream>>>(xgcn, ri, ro, gb3, xgcn, nullptr, N, OUTD, OUTD, 0);

    // ---- MLP head ----
    k_concat_bf<<<g1((long)N * 128), 256, 0, stream>>>(rst2, xgcn, zbf, N);
    run_gemm(zbf, f1wt, z1f, z1bf, f1b, N, 128, 64, 64, 64, /*leaky*/2, stream);
    run_gemm(z1bf, f2wt, (float*)d_out, nullptr, f2b, N, 64, 2, 2, 0, 0, stream);
}
